// gcn_nifa_39367670235688
// MI455X (gfx1250) — matrix operand order repaired and checked by reading, not compiled, not run on hardware
//
#include <hip/hip_runtime.h>

typedef _Float16 v16h __attribute__((ext_vector_type(16)));
typedef _Float16 v8h  __attribute__((ext_vector_type(8)));
typedef _Float16 v4h  __attribute__((ext_vector_type(4)));
typedef float    v8f  __attribute__((ext_vector_type(8)));

// hardware fp32 atomic add (global_atomic_add_f32), relaxed + agent scope
__device__ __forceinline__ void atomAddF(float* p, float v) {
    __hip_atomic_fetch_add(p, v, __ATOMIC_RELAXED, __HIP_MEMORY_SCOPE_AGENT);
}

// ---------------- utility kernels ----------------

__global__ __launch_bounds__(256) void k_zero(float* __restrict__ p, long n4) {
    long i = (long)blockIdx.x * 256 + threadIdx.x;
    if (i < n4) ((float4*)p)[i] = make_float4(0.f, 0.f, 0.f, 0.f);
}

__global__ __launch_bounds__(256) void k_degree(const int* __restrict__ src,
                                                const int* __restrict__ dst,
                                                float* __restrict__ degS,
                                                float* __restrict__ degD, int E) {
    int e = blockIdx.x * 256 + threadIdx.x;
    if (e < E) {
        atomAddF(&degS[src[e]], 1.0f);
        atomAddF(&degD[dst[e]], 1.0f);
    }
}

__global__ __launch_bounds__(256) void k_isqrt(float* __restrict__ a,
                                               float* __restrict__ b, int n) {
    int i = blockIdx.x * 256 + threadIdx.x;
    if (i < n) {
        a[i] = rsqrtf(fmaxf(a[i], 1.0f));
        b[i] = rsqrtf(fmaxf(b[i], 1.0f));
    }
}

// ---------------- layer 1 GEMM: H1 = (X @ W1) * isqS, via WMMA f16 ----------------
// block = 256 threads = 8 waves; wave computes 16 rows x 128 cols.
// W1 (256x128 f32) is staged transposed+f16 in LDS: Wt[col][k], stride 264 halves.

__global__ __launch_bounds__(256) void k_gemm1(const float* __restrict__ X,
                                               const float* __restrict__ W1,
                                               const float* __restrict__ isqS,
                                               float* __restrict__ H1, int n) {
    __shared__ _Float16 Bl[128 * 264]; // 67,584 B (CDNA5: 320KB/WGP)
    for (int idx = threadIdx.x; idx < 256 * 128; idx += 256) {
        int k = idx >> 7, c = idx & 127;
        Bl[c * 264 + k] = (_Float16)W1[idx];
    }
    __syncthreads();

    const int wave = threadIdx.x >> 5, lane = threadIdx.x & 31;
    const int r0 = (blockIdx.x * 8 + wave) * 16;
    if (r0 >= n) return;                       // wave-uniform: EXEC stays all-1s
    const int lrow = lane & 15;
    const int hi   = lane >> 4;
    const int row  = r0 + lrow;
    const int ka   = hi ? 8 : 0;               // A-matrix K sub-offset
    const int kb   = hi ? 8 : 0;               // B-matrix K sub-offset

    v8f acc[8] = {};
    const float* xrow = X + (long)row * 256;

    for (int kk = 0; kk < 256; kk += 32) {
        float4 f0 = *(const float4*)(xrow + kk + ka);
        float4 f1 = *(const float4*)(xrow + kk + ka + 4);
        float4 f2 = *(const float4*)(xrow + kk + ka + 16);
        float4 f3 = *(const float4*)(xrow + kk + ka + 20);
        v16h a;
        a[0]=(_Float16)f0.x;  a[1]=(_Float16)f0.y;  a[2]=(_Float16)f0.z;  a[3]=(_Float16)f0.w;
        a[4]=(_Float16)f1.x;  a[5]=(_Float16)f1.y;  a[6]=(_Float16)f1.z;  a[7]=(_Float16)f1.w;
        a[8]=(_Float16)f2.x;  a[9]=(_Float16)f2.y;  a[10]=(_Float16)f2.z; a[11]=(_Float16)f2.w;
        a[12]=(_Float16)f3.x; a[13]=(_Float16)f3.y; a[14]=(_Float16)f3.z; a[15]=(_Float16)f3.w;
#pragma unroll
        for (int t = 0; t < 8; ++t) {
            const _Float16* bp = &Bl[(t * 16 + lrow) * 264 + kk + kb];
            v8h b0 = *(const v8h*)bp;
            v8h b1 = *(const v8h*)(bp + 16);
            v16h b;
#pragma unroll
            for (int j = 0; j < 8; ++j) { b[j] = b0[j]; b[8 + j] = b1[j]; }
            acc[t] = __builtin_amdgcn_wmma_f32_16x16x32_f16(
                false, a, false, b, (short)0, acc[t], false, false);
        }
    }
    // epilogue: scale by out-degree norm, store fp32
#pragma unroll
    for (int i = 0; i < 8; ++i) {
        int m = r0 + i + (hi ? 8 : 0);
        float s = isqS[m];
        float* hp = H1 + (long)m * 128 + lrow;
#pragma unroll
        for (int t = 0; t < 8; ++t) hp[t * 16] = acc[t][i] * s;
    }
}

// ---------------- layer 1 aggregation: M1[dst] += H1[src], warp per edge ----------------

__global__ __launch_bounds__(256) void k_agg1(const int* __restrict__ src,
                                              const int* __restrict__ dst,
                                              const float* __restrict__ H1,
                                              float* __restrict__ M1, int E) {
    int t = blockIdx.x * 256 + threadIdx.x;
    int e = t >> 5, lane = t & 31;
    if (e >= E) return;
    int s = src[e], d = dst[e];
    const float4 v = *(const float4*)(H1 + (long)s * 128 + lane * 4);
    float* mp = M1 + (long)d * 128 + lane * 4;
    atomAddF(mp + 0, v.x); atomAddF(mp + 1, v.y);
    atomAddF(mp + 2, v.z); atomAddF(mp + 3, v.w);
}

// ---------------- norm + bias + ReLU, convert to f16 for layer-2 GEMM ----------------

__global__ __launch_bounds__(256) void k_relu(const float* __restrict__ M1,
                                              const float* __restrict__ isqD,
                                              const float* __restrict__ b1,
                                              _Float16* __restrict__ H1R, int n) {
    int i = blockIdx.x * 256 + threadIdx.x;   // float4 index
    if (i >= n * 32) return;
    int idx = i * 4;
    int row = idx >> 7, col = idx & 127;
    float s = isqD[row];
    float4 v  = *(const float4*)(M1 + idx);
    float4 bb = *(const float4*)(b1 + col);
    v4h h;
    h[0] = (_Float16)fmaxf(v.x * s + bb.x, 0.f);
    h[1] = (_Float16)fmaxf(v.y * s + bb.y, 0.f);
    h[2] = (_Float16)fmaxf(v.z * s + bb.z, 0.f);
    h[3] = (_Float16)fmaxf(v.w * s + bb.w, 0.f);
    *(v4h*)(H1R + idx) = h;
}

// ---------------- layer 2 GEMM: Z2 = (H1R @ W2) * isqS, cols padded 40->48 ----------------

__global__ __launch_bounds__(256) void k_gemm2(const _Float16* __restrict__ A,
                                               const float* __restrict__ W2,
                                               const float* __restrict__ isqS,
                                               float* __restrict__ Z2, int n) {
    __shared__ _Float16 Bl[48 * 136]; // 13,056 B
    for (int idx = threadIdx.x; idx < 128 * 48; idx += 256) {
        int k = idx / 48, c = idx - k * 48;
        float v = (c < 40) ? W2[k * 40 + c] : 0.f;
        Bl[c * 136 + k] = (_Float16)v;
    }
    __syncthreads();

    const int wave = threadIdx.x >> 5, lane = threadIdx.x & 31;
    const int r0 = (blockIdx.x * 8 + wave) * 16;
    if (r0 >= n) return;
    const int lrow = lane & 15, hi = lane >> 4;
    const int row = r0 + lrow;
    const int ka = hi ? 8 : 0, kb = hi ? 8 : 0;

    v8f acc[3] = {};
    const _Float16* arow = A + (long)row * 128;
#pragma unroll
    for (int kk = 0; kk < 128; kk += 32) {
        v8h a0 = *(const v8h*)(arow + kk + ka);
        v8h a1 = *(const v8h*)(arow + kk + ka + 16);
        v16h a;
#pragma unroll
        for (int j = 0; j < 8; ++j) { a[j] = a0[j]; a[8 + j] = a1[j]; }
#pragma unroll
        for (int t = 0; t < 3; ++t) {
            const _Float16* bp = &Bl[(t * 16 + lrow) * 136 + kk + kb];
            v8h b0 = *(const v8h*)bp;
            v8h b1 = *(const v8h*)(bp + 16);
            v16h b;
#pragma unroll
            for (int j = 0; j < 8; ++j) { b[j] = b0[j]; b[8 + j] = b1[j]; }
            acc[t] = __builtin_amdgcn_wmma_f32_16x16x32_f16(
                false, a, false, b, (short)0, acc[t], false, false);
        }
    }
#pragma unroll
    for (int i = 0; i < 8; ++i) {
        int m = r0 + i + (hi ? 8 : 0);
        float s = isqS[m];
        float* zp = Z2 + (long)m * 48 + lrow;
#pragma unroll
        for (int t = 0; t < 3; ++t) zp[t * 16] = acc[t][i] * s;
    }
}

// ---------------- layer 2 aggregation: 40 useful cols (stride 48), warp per edge ----------------

__global__ __launch_bounds__(256) void k_agg2(const int* __restrict__ src,
                                              const int* __restrict__ dst,
                                              const float* __restrict__ Z2,
                                              float* __restrict__ M2, int E) {
    int t = blockIdx.x * 256 + threadIdx.x;
    int e = t >> 5, lane = t & 31;
    if (e >= E) return;
    int s = src[e], d = dst[e];
    const float* zp = Z2 + (long)s * 48;
    float* mp = M2 + (long)d * 48;
    atomAddF(mp + lane, zp[lane]);
    if (lane < 8) atomAddF(mp + 32 + lane, zp[32 + lane]);
}

// ---------------- final: out = M2 * isqD + b2 (fp32, [n,40]) ----------------

__global__ __launch_bounds__(256) void k_final(const float* __restrict__ M2,
                                               const float* __restrict__ isqD,
                                               const float* __restrict__ b2,
                                               float* __restrict__ out, int n) {
    int i = blockIdx.x * 256 + threadIdx.x;
    if (i >= n * 40) return;
    int row = i / 40, col = i - row * 40;
    out[i] = M2[(long)row * 48 + col] * isqD[row] + b2[col];
}

// ---------------- host-side orchestration ----------------

extern "C" void kernel_launch(void* const* d_in, const int* in_sizes, int n_in,
                              void* d_out, int out_size, void* d_ws, size_t ws_size,
                              hipStream_t stream) {
    const float* X  = (const float*)d_in[0];
    const int*   src = (const int*)d_in[1];
    const int*   dst = (const int*)d_in[2];
    const float* W1 = (const float*)d_in[3];
    const float* b1 = (const float*)d_in[4];
    const float* W2 = (const float*)d_in[5];
    const float* b2 = (const float*)d_in[6];
    float* out = (float*)d_out;

    const int n = in_sizes[0] / 256;   // 100000
    const int E = in_sizes[1];         // 1600000
    const int NP = (n + 63) & ~63;     // padded (keeps 256B-aligned ws offsets)

    char* ws = (char*)d_ws;
    size_t off = 0;
    auto carve = [&](size_t bytes) -> char* {
        char* p = ws + off;
        off = (off + bytes + 255) & ~(size_t)255;
        return p;
    };
    float*    isqS = (float*)carve((size_t)NP * 4);
    float*    isqD = (float*)carve((size_t)NP * 4);       // contiguous after isqS
    float*    H1   = (float*)carve((size_t)n * 128 * 4);
    float*    M1   = (float*)carve((size_t)n * 128 * 4);
    _Float16* H1R  = (_Float16*)carve((size_t)n * 128 * 2);
    float*    Z2   = (float*)carve((size_t)n * 48 * 4);
    float*    M2   = (float*)carve((size_t)n * 48 * 4);
    (void)n_in; (void)out_size; (void)ws_size;

    // zero degree arrays (contiguous pair) and accumulators each call
    long z0 = (long)NP * 2 / 4;
    k_zero<<<(unsigned)((z0 + 255) / 256), 256, 0, stream>>>(isqS, z0);
    long z1 = (long)n * 128 / 4;
    k_zero<<<(unsigned)((z1 + 255) / 256), 256, 0, stream>>>(M1, z1);
    long z2 = (long)n * 48 / 4;
    k_zero<<<(unsigned)((z2 + 255) / 256), 256, 0, stream>>>(M2, z2);

    k_degree<<<(E + 255) / 256, 256, 0, stream>>>(src, dst, isqS, isqD, E);
    k_isqrt<<<(n + 255) / 256, 256, 0, stream>>>(isqS, isqD, n);

    k_gemm1<<<(n + 127) / 128, 256, 0, stream>>>(X, W1, isqS, H1, n);

    long th = (long)E * 32;
    k_agg1<<<(unsigned)((th + 255) / 256), 256, 0, stream>>>(src, dst, H1, M1, E);

    k_relu<<<(unsigned)(((long)n * 32 + 255) / 256), 256, 0, stream>>>(M1, isqD, b1, H1R, n);

    k_gemm2<<<(n + 127) / 128, 256, 0, stream>>>(H1R, W2, isqS, Z2, n);

    k_agg2<<<(unsigned)((th + 255) / 256), 256, 0, stream>>>(src, dst, Z2, M2, E);

    k_final<<<(unsigned)(((long)n * 40 + 255) / 256), 256, 0, stream>>>(M2, isqD, b2, out, n);
}
